// GCN_45002667327959
// MI455X (gfx1250) — matrix operand order repaired and checked by reading, not compiled, not run on hardware
//
#include <hip/hip_runtime.h>
#include <hip/hip_bf16.h>

#define E      2048
#define IN_DIM 32
#define HEADS  3
#define DHEAD  6
#define HID    18
#define NCOMBO 6     // 2 gats x 3 heads
#define TI     256   // i-chunk (rows) staged per TDM load in k_gat_aggr

typedef __attribute__((ext_vector_type(16))) __bf16   v16bf;
typedef __attribute__((ext_vector_type(8)))  float    v8f;
typedef __attribute__((ext_vector_type(4)))  float    v4f;
typedef __attribute__((ext_vector_type(4)))  unsigned v4u;
typedef __attribute__((ext_vector_type(8)))  unsigned v8u;

// Branchless GAT logit: adj in {0,1} ->  a ? leaky_0.2(v) : -30
__device__ __forceinline__ float gat_logit(float a, float v)
{
    float lv = fmaxf(v, 0.2f * v);               // LeakyReLU, slope 0.2 < 1
    return __builtin_fmaf(a, lv + 30.f, -30.f);  // a*(lv+30) - 30
}

// ---------------------------------------------------------------------------
// Kernel A: per-node features for batch 0: h = x@W, e1 = h.a1, e2 = h.a2.
// Stores h transposed in bf16, padded to 16 rows (rows 6..15 = 0) per combo.
// ---------------------------------------------------------------------------
__global__ void k_feat(const float* __restrict__ x0, const float* __restrict__ x1,
                       const float* __restrict__ Wt1, const float* __restrict__ at1,
                       const float* __restrict__ Wt2, const float* __restrict__ at2,
                       __bf16* __restrict__ hT, float* __restrict__ e1g,
                       float* __restrict__ e2g)
{
    int t = blockIdx.x * blockDim.x + threadIdx.x;
    if (t >= 2 * E) return;
    int gat = t / E, i = t % E;
    const float* x = (gat == 0 ? x0 : x1) + (size_t)i * IN_DIM;
    const float* W = (gat == 0 ? Wt1 : Wt2);
    const float* a = (gat == 0 ? at1 : at2);
    float xr[IN_DIM];
#pragma unroll
    for (int k = 0; k < IN_DIM; ++k) xr[k] = x[k];
    for (int hh = 0; hh < HEADS; ++hh) {
        float h[DHEAD];
#pragma unroll
        for (int d = 0; d < DHEAD; ++d) {
            float s = 0.f;
#pragma unroll
            for (int k = 0; k < IN_DIM; ++k)
                s += xr[k] * W[hh * IN_DIM * DHEAD + k * DHEAD + d];
            h[d] = s;
        }
        float e1 = 0.f, e2 = 0.f;
#pragma unroll
        for (int d = 0; d < DHEAD; ++d) {
            e1 += h[d] * a[hh * 2 * DHEAD + d];
            e2 += h[d] * a[hh * 2 * DHEAD + DHEAD + d];
        }
        int c = gat * HEADS + hh;
#pragma unroll
        for (int d = 0; d < DHEAD; ++d)
            hT[(size_t)(c * 16 + d) * E + i] = (__bf16)h[d];
#pragma unroll
        for (int d = DHEAD; d < 16; ++d)
            hT[(size_t)(c * 16 + d) * E + i] = (__bf16)0.f;
        e1g[c * E + i] = e1;
        e2g[c * E + i] = e2;
    }
}

// ---------------------------------------------------------------------------
// Kernel B: softmax denominators. One wave per adjacency row i, all 6 combos
// in one pass over adj0[i,:]. Stores reciprocal 1/sum_j exp(logit[i,j]).
// ---------------------------------------------------------------------------
__global__ void k_denom(const float* __restrict__ adj0, const float* __restrict__ e1g,
                        const float* __restrict__ e2g, float* __restrict__ rcg)
{
    int wid  = (blockIdx.x * blockDim.x + threadIdx.x) >> 5;
    int lane = threadIdx.x & 31;
    if (wid >= E) return;
    int i = wid;
    float e1[NCOMBO];
#pragma unroll
    for (int c = 0; c < NCOMBO; ++c) e1[c] = e1g[c * E + i];
    float s[NCOMBO] = {0.f, 0.f, 0.f, 0.f, 0.f, 0.f};
    for (int j = lane; j < E; j += 32) {
        float a = adj0[(size_t)i * E + j];
#pragma unroll
        for (int c = 0; c < NCOMBO; ++c) {
            float v = e1[c] + e2g[c * E + j];
            s[c] += __expf(gat_logit(a, v));
        }
    }
#pragma unroll
    for (int c = 0; c < NCOMBO; ++c) {
        float v = s[c];
        for (int off = 16; off > 0; off >>= 1) v += __shfl_xor(v, off, 32);
        if (lane == 0) rcg[c * E + i] = 1.f / v;
    }
}

// ---------------------------------------------------------------------------
// Kernel C: fused attention + aggregation, h'[j,d] = sum_i att[i,j]*h[i,d].
// One block per 16-wide j tile; 6 waves, one (gat,head) combo each. The adj
// tile (TI x 16 f32) is DMA'd into LDS by the Tensor Data Mover (one
// tensor_load_to_lds per chunk, TENSORcnt-tracked), attention tiles are built
// branchlessly in registers per the WMMA 16-bit A layout and fed to
// v_wmma_f32_16x16x32_bf16.
// ---------------------------------------------------------------------------
__global__ void __launch_bounds__(192) k_gat_aggr(
    const float* __restrict__ adj0, const __bf16* __restrict__ hT,
    const float* __restrict__ e1g, const float* __restrict__ e2g,
    const float* __restrict__ rcg, float* __restrict__ hp)
{
    __shared__ float lds_adj[TI][16];     // [i-row][j-col], row-major as TDM writes it
    __shared__ float lds_e1[NCOMBO][TI];
    __shared__ float lds_rc[NCOMBO][TI];

    const int j0   = blockIdx.x * 16;
    const int tid  = threadIdx.x;
    const int wave = tid >> 5;            // 0..5 == combo
    const int lane = tid & 31;
    const int c    = wave;
    const int m    = lane & 15;           // A row (j), B/D column (d)
    const int hi   = lane >> 4;
    const int kb8  = hi * 8;
    const int kb16 = hi * 16;

    const float e2j = e2g[c * E + j0 + m];
    v8f acc = {};

    for (int i0 = 0; i0 < E; i0 += TI) {
        __syncthreads();
        // stage e1 / recip chunk cooperatively
        for (int idx = tid; idx < NCOMBO * TI; idx += 192) {
            int cc = idx / TI, l = idx - cc * TI;
            lds_e1[cc][l] = e1g[cc * E + i0 + l];
            lds_rc[cc][l] = rcg[cc * E + i0 + l];
        }
        if (wave == 0) {
            // Tensor DMA descriptor: 2D tile TI rows x 16 cols of f32 from a
            // 2048x2048 tensor, stride 2048, into lds_adj (row-major).
            unsigned lds_off = (unsigned)(size_t)(&lds_adj[0][0]);
            unsigned long long ga =
                (unsigned long long)(size_t)adj0 + ((size_t)i0 * E + j0) * 4ull;
            v4u g0 = { 1u,                                  // count=1
                       lds_off,                             // lds_addr
                       (unsigned)(ga & 0xFFFFFFFFu),        // global_addr lo
                       (((unsigned)(ga >> 32)) & 0x1FFFFFFu) | (2u << 30) }; // hi|type=2
            v8u g1 = { 0x20000u,                            // data_size=4B
                       (unsigned)E << 16,                   // tensor_dim0 lo16
                       (unsigned)E << 16,                   // dim0 hi|tensor_dim1 lo16
                       16u << 16,                           // dim1 hi | tile_dim0=16
                       (unsigned)TI,                        // tile_dim1=TI, tile_dim2=0
                       (unsigned)E,                         // tensor_dim0_stride lo32
                       0u, 0u };
            asm volatile("tensor_load_to_lds %0, %1" :: "s"(g0), "s"(g1) : "memory");
            __builtin_amdgcn_s_wait_tensorcnt(0);
        }
        __syncthreads();

        for (int k8 = 0; k8 < TI; k8 += 32) {
            const int kbase = k8 + kb8;
            // e1 / recip for this lane's 16 K slots: two contiguous 8-runs
            float e1v[16], rcv[16];
            *(v4f*)&e1v[0]  = *(const v4f*)&lds_e1[c][kbase];
            *(v4f*)&e1v[4]  = *(const v4f*)&lds_e1[c][kbase + 4];
            *(v4f*)&e1v[8]  = *(const v4f*)&lds_e1[c][kbase + 16];
            *(v4f*)&e1v[12] = *(const v4f*)&lds_e1[c][kbase + 20];
            *(v4f*)&rcv[0]  = *(const v4f*)&lds_rc[c][kbase];
            *(v4f*)&rcv[4]  = *(const v4f*)&lds_rc[c][kbase + 4];
            *(v4f*)&rcv[8]  = *(const v4f*)&lds_rc[c][kbase + 16];
            *(v4f*)&rcv[12] = *(const v4f*)&lds_rc[c][kbase + 20];

            v16bf A;
#pragma unroll
            for (int e = 0; e < 16; ++e) {
                int   k = kbase + (e < 8 ? e : e + 8);
                float a = lds_adj[k][m];
                A[e] = (__bf16)(__expf(gat_logit(a, e1v[e] + e2j)) * rcv[e]);
            }
            v16bf B;
#pragma unroll
            for (int e = 0; e < 16; ++e)
                B[e] = hT[(size_t)(c * 16 + m) * E + i0 + k8 + kb8 + (e < 8 ? e : e + 8)];
            acc = __builtin_amdgcn_wmma_f32_16x16x32_bf16(false, A, false, B,
                                                          (short)0, acc, false, false);
        }
    }

    if (m < DHEAD) {
        int gat = c / HEADS, hh = c % HEADS;
        int f = hh * DHEAD + m;
#pragma unroll
        for (int r = 0; r < 8; ++r) {
            int   j = j0 + r + hi * 8;
            float v = acc[r];
            v = (v > 0.f) ? v : (__expf(v) - 1.f);          // ELU
            hp[(size_t)gat * E * HID + (size_t)j * HID + f] = v;
        }
    }
}

// ---------------------------------------------------------------------------
// Kernel D: T1 = hp @ W1 (18x16), stored bf16-transposed per stream.
// ---------------------------------------------------------------------------
__global__ void k_lin1(const float* __restrict__ hp, const float* __restrict__ Wg1,
                       const float* __restrict__ Wo1, __bf16* __restrict__ T1T)
{
    int t = blockIdx.x * blockDim.x + threadIdx.x;
    if (t >= 2 * E) return;
    int s = t / E, i = t % E;
    const float* W   = s ? Wo1 : Wg1;                  // [18,16] row-major
    const float* row = hp + (size_t)s * E * HID + (size_t)i * HID;
#pragma unroll
    for (int ccol = 0; ccol < 16; ++ccol) {
        float a = 0.f;
#pragma unroll
        for (int k = 0; k < HID; ++k) a += row[k] * W[k * 16 + ccol];
        T1T[(size_t)(s * 16 + ccol) * E + i] = (__bf16)a;
    }
}

// ---------------------------------------------------------------------------
// Kernel E: out = [relu](adj0 @ B + bias) — LDS-free WMMA GEMM computing the
// transposed product: D[col][row] = sum_k BT[col][k] * adj[row][k]. Both
// operands stream contiguously from global (A = bf16 rows of BT, B = f32 rows
// of adj converted in-register). 2 waves/block, one per stream.
// ---------------------------------------------------------------------------
__global__ void __launch_bounds__(64) k_adjmm(
    const float* __restrict__ adj0, const __bf16* __restrict__ BT,
    const float* __restrict__ bias0, const float* __restrict__ bias1,
    int ncols, int do_relu, float* __restrict__ out, int out_stride)
{
    const int tid  = threadIdx.x;
    const int s    = tid >> 5;          // stream
    const int lane = tid & 31;
    const int l16  = lane & 15;         // A row (out col) and B col (out row)
    const int hi   = lane >> 4;
    const int kb8  = hi * 8;
    const int kb16 = hi * 16;
    const int m0   = blockIdx.x * 16;

    const __bf16* pa_base = BT + (size_t)(s * 16 + l16) * E;
    const float*  pb_base = adj0 + (size_t)(m0 + l16) * E;

    v8f acc = {};
    for (int k0 = 0; k0 < E; k0 += 32) {
        const __bf16* pa = pa_base + k0 + kb16;
        const float*  pb = pb_base + k0 + kb16;
        v16bf A, B;
#pragma unroll
        for (int e = 0; e < 16; ++e) {
            A[e] = pa[e];
            B[e] = (__bf16)pb[e];
        }
        acc = __builtin_amdgcn_wmma_f32_16x16x32_bf16(false, A, false, B,
                                                      (short)0, acc, false, false);
    }

    const float* bias = s ? bias1 : bias0;
    const int mrow = m0 + l16;
#pragma unroll
    for (int r = 0; r < 8; ++r) {
        int col = r + hi * 8;
        if (col < ncols) {
            float v = acc[r] + bias[col];
            if (do_relu) v = fmaxf(v, 0.f);
            out[(size_t)s * E * out_stride + (size_t)mrow * out_stride + col] = v;
        }
    }
}

// ---------------------------------------------------------------------------
// Kernel F: T2 = Z1 @ W2 (16x2), stored bf16-transposed, rows 2..15 zeroed.
// ---------------------------------------------------------------------------
__global__ void k_lin2(const float* __restrict__ Z1, const float* __restrict__ Wg2,
                       const float* __restrict__ Wo2, __bf16* __restrict__ T2T)
{
    int t = blockIdx.x * blockDim.x + threadIdx.x;
    if (t >= 2 * E) return;
    int s = t / E, i = t % E;
    const float* W   = s ? Wo2 : Wg2;                  // [16,2] row-major
    const float* row = Z1 + (size_t)s * E * 16 + (size_t)i * 16;
#pragma unroll
    for (int ccol = 0; ccol < 2; ++ccol) {
        float a = 0.f;
#pragma unroll
        for (int k = 0; k < 16; ++k) a += row[k] * W[k * 2 + ccol];
        T2T[(size_t)(s * 16 + ccol) * E + i] = (__bf16)a;
    }
#pragma unroll
    for (int ccol = 2; ccol < 16; ++ccol)
        T2T[(size_t)(s * 16 + ccol) * E + i] = (__bf16)0.f;
}

// ---------------------------------------------------------------------------
// Kernel H: log_softmax over the 2 classes + LeakyReLU(0.01).
// ---------------------------------------------------------------------------
__global__ void k_final(const float* __restrict__ Z2, float* __restrict__ out)
{
    int i = blockIdx.x * blockDim.x + threadIdx.x;
    if (i >= E) return;
    float y0 = Z2[(size_t)i * 2], y1 = Z2[(size_t)i * 2 + 1];
    float mx = fmaxf(y0, y1);
    float l  = __logf(__expf(y0 - mx) + __expf(y1 - mx));
    out[(size_t)i * 2]     = y0 - mx - l;
    out[(size_t)i * 2 + 1] = y1 - mx - l;
    float o0 = Z2[(size_t)E * 2 + (size_t)i * 2];
    float o1 = Z2[(size_t)E * 2 + (size_t)i * 2 + 1];
    out[(size_t)E * 2 + (size_t)i * 2]     = o0 > 0.f ? o0 : 0.01f * o0;
    out[(size_t)E * 2 + (size_t)i * 2 + 1] = o1 > 0.f ? o1 : 0.01f * o1;
}

// ---------------------------------------------------------------------------
extern "C" void kernel_launch(void* const* d_in, const int* in_sizes, int n_in,
                              void* d_out, int out_size, void* d_ws, size_t ws_size,
                              hipStream_t stream)
{
    const float* node = (const float*)d_in[0];   // [4,2048,32] -> batch 0
    const float* uvf  = (const float*)d_in[1];
    const float* adj0 = (const float*)d_in[2];   // [4,2048,2048] -> batch 0
    const float* Wt1  = (const float*)d_in[3];
    const float* at1  = (const float*)d_in[4];
    const float* Wt2  = (const float*)d_in[5];
    const float* at2  = (const float*)d_in[6];
    const float* Wg1  = (const float*)d_in[7];
    const float* bg1  = (const float*)d_in[8];
    const float* Wg2  = (const float*)d_in[9];
    const float* bg2  = (const float*)d_in[10];
    const float* Wo1  = (const float*)d_in[11];
    const float* bo1  = (const float*)d_in[12];
    const float* Wo2  = (const float*)d_in[13];
    const float* bo2  = (const float*)d_in[14];

    char* w = (char*)d_ws;
    __bf16* hT  = (__bf16*)(w + 0);         // 6*16*2048*2  = 393216
    float*  e1g = (float*)(w + 393216);     // 6*2048*4     =  49152
    float*  e2g = (float*)(w + 442368);
    float*  rcg = (float*)(w + 491520);
    float*  hp  = (float*)(w + 540672);     // 2*2048*18*4  = 294912
    __bf16* T1T = (__bf16*)(w + 835584);    // 2*16*2048*2  = 131072
    float*  Z1  = (float*)(w + 966656);     // 2*2048*16*4  = 262144
    __bf16* T2T = (__bf16*)(w + 1228800);   // 2*16*2048*2  = 131072
    float*  Z2  = (float*)(w + 1359872);    // 2*2048*2*4   =  32768

    k_feat<<<(2 * E + 255) / 256, 256, 0, stream>>>(node, uvf, Wt1, at1, Wt2, at2,
                                                    hT, e1g, e2g);
    k_denom<<<E / 8, 256, 0, stream>>>(adj0, e1g, e2g, rcg);
    k_gat_aggr<<<E / 16, 192, 0, stream>>>(adj0, hT, e1g, e2g, rcg, hp);
    k_lin1<<<(2 * E + 255) / 256, 256, 0, stream>>>(hp, Wg1, Wo1, T1T);
    k_adjmm<<<E / 16, 64, 0, stream>>>(adj0, T1T, bg1, bo1, 16, 1, Z1, 16);
    k_lin2<<<(2 * E + 255) / 256, 256, 0, stream>>>(Z1, Wg2, Wo2, T2T);
    k_adjmm<<<E / 16, 64, 0, stream>>>(adj0, T2T, bg2, bo2, 2, 0, Z2, 2);
    k_final<<<(E + 255) / 256, 256, 0, stream>>>(Z2, (float*)d_out);
}
